// MultiScaleEdgeGATLayer_45947560133207
// MI455X (gfx1250) — matrix operand order repaired and checked by reading, not compiled, not run on hardware
//
#include <hip/hip_runtime.h>
#include <math.h>

#define N_NODES 100000
#define N_EDGES 1600000
#define IN_DIM 128
#define OUT_DIM 128
#define EDGE_DIM 32
#define HEADS 4
#define HEAD_DIM 32
#define NEG_SLOPE 0.2f
#define LN_EPS 1e-5f

typedef __attribute__((ext_vector_type(16))) _Float16 v16h;
typedef __attribute__((ext_vector_type(8)))  float    v8f;

__device__ __forceinline__ v8f wmma_f16(v16h a, v16h b, v8f c) {
    // D = A(16x32 f16) * B(32x16 f16) + C(16x16 f32)
    return __builtin_amdgcn_wmma_f32_16x16x32_f16(
        /*neg_a=*/false, a, /*neg_b=*/false, b,
        /*c_mod=*/(short)0, c, /*reuse_a=*/false, /*reuse_b=*/false);
}

// A-fragment K index for element h (CDNA5 16-bit A 16x32 layout, §7.12.2):
// VGPR v=h/2 holds 2 halfs; K = (v/4)*16 + (lane/16)*8 + (v%4)*2 + (h%2)
__device__ __forceinline__ int a_kidx(int h, int g) {
    int v = h >> 1, p = h & 1;
    return ((v >> 2) << 4) + (g << 3) + ((v & 3) << 1) + p;
}

__device__ __forceinline__ void atomicMaxF(float* addr, float val) {
    if (val >= 0.0f) atomicMax((int*)addr, __float_as_int(val));
    else             atomicMin((unsigned int*)addr, __float_as_uint(val));
}

// ---------------- pack W[O x K] (row-major, O==128) into f16 B-frag layout ---
// packed[(((ct*KC)+kc)*32 + lane)*16 + s] = W[ct*16+lane%16][kc*32+a_kidx(s,lane/16)]
// Row o < split comes from W0, else from W1 (o-split). KC = K/32.
__global__ void k_pack_w(const float* __restrict__ W0, const float* __restrict__ W1,
                         int split, int K, _Float16* __restrict__ out, int n_elems) {
    int i = blockIdx.x * blockDim.x + threadIdx.x;
    if (i >= n_elems) return;
    const int KC = K >> 5;
    const int nl   = i & 15;
    const int kl   = (i >> 4) & 31;
    const int rest = i >> 9;
    const int kc   = rest % KC;
    const int ct   = rest / KC;
    const int o = ct * 16 + (kl & 15);
    const int k = kc * 32 + a_kidx(nl, kl >> 4);
    const float v = (o < split) ? W0[(size_t)o * K + k] : W1[(size_t)(o - split) * K + k];
    out[i] = (_Float16)v;
}

// ---------------- init: zero accumulators, mx = -inf ----------------
__global__ void k_init(float* __restrict__ hacc, float* __restrict__ eagg,
                       float* __restrict__ denom, float* __restrict__ mx,
                       float* __restrict__ deg) {
    long long i = (long long)blockIdx.x * blockDim.x + threadIdx.x;
    if (i < (long long)N_NODES * OUT_DIM) { hacc[i] = 0.0f; eagg[i] = 0.0f; }
    if (i < (long long)N_NODES * HEADS)   { denom[i] = 0.0f; mx[i] = -__builtin_inff(); }
    if (i < N_NODES)                      { deg[i] = 0.0f; }
}

// ---------------- fs/fd = X @ W^T + b  (WMMA; full 16x128 row tile / wave) ---
__global__ void __launch_bounds__(32)
k_node_linear(const float* __restrict__ x,
              const _Float16* __restrict__ pWsrc, const float* __restrict__ bsrc,
              const _Float16* __restrict__ pWdst, const float* __restrict__ bdst,
              float* __restrict__ fs, float* __restrict__ fd) {
    const int lane = threadIdx.x & 31;
    const int m = lane & 15, g = lane >> 4;
    const int nt = blockIdx.x * 16;        // node tile
    const v16h* __restrict__ pB = (const v16h*)(blockIdx.y ? pWdst : pWsrc);
    const float* __restrict__ bias = blockIdx.y ? bdst : bsrc;
    float* __restrict__ out        = blockIdx.y ? fd : fs;

    // A fragments for all 4 K-chunks (kept in registers, reused across 8 col tiles)
    v16h a[4];
    const float* xrow = x + (size_t)(nt + m) * IN_DIM;
    #pragma unroll
    for (int kc = 0; kc < IN_DIM / 32; ++kc)
        #pragma unroll
        for (int h = 0; h < 16; ++h)
            a[kc][h] = (_Float16)xrow[kc * 32 + a_kidx(h, g)];

    #pragma unroll
    for (int ct = 0; ct < OUT_DIM / 16; ++ct) {
        v8f c = {};
        #pragma unroll
        for (int kc = 0; kc < IN_DIM / 32; ++kc)
            c = wmma_f16(a[kc], pB[(ct * (IN_DIM / 32) + kc) * 32 + lane], c);
        const int n = ct * 16 + m;
        const float bv = bias[n];
        #pragma unroll
        for (int r = 0; r < 8; ++r)        // C[r] = row (r + 8g), col m
            out[(size_t)(nt + r + g * 8) * OUT_DIM + n] = c[r] + bv;
    }
}

// ---------------- attention logits + segment max + degree --------------------
__global__ void k_edge_logits(const float* __restrict__ fs, const float* __restrict__ fd,
                              const int* __restrict__ src, const int* __restrict__ dst,
                              const float* __restrict__ attn_w,
                              float* __restrict__ albuf, float* __restrict__ mx,
                              float* __restrict__ deg) {
    long long t = (long long)blockIdx.x * blockDim.x + threadIdx.x;
    if (t >= (long long)N_EDGES * HEADS) return;
    const int e = (int)(t >> 2), hh = (int)(t & 3);
    const int s = src[e], d = dst[e];
    const float4* ps = (const float4*)(fs + (size_t)s * OUT_DIM + hh * HEAD_DIM);
    const float4* pd = (const float4*)(fd + (size_t)d * OUT_DIM + hh * HEAD_DIM);
    const float4* pw = (const float4*)(attn_w + hh * HEAD_DIM);
    float acc = 0.0f;
    #pragma unroll
    for (int j = 0; j < HEAD_DIM / 4; ++j) {
        float4 a4 = ps[j], b4 = pd[j], w4 = pw[j];
        float v;
        v = a4.x + b4.x; v = v > 0.0f ? v : NEG_SLOPE * v; acc += v * w4.x;
        v = a4.y + b4.y; v = v > 0.0f ? v : NEG_SLOPE * v; acc += v * w4.y;
        v = a4.z + b4.z; v = v > 0.0f ? v : NEG_SLOPE * v; acc += v * w4.z;
        v = a4.w + b4.w; v = v > 0.0f ? v : NEG_SLOPE * v; acc += v * w4.w;
    }
    albuf[(size_t)e * HEADS + hh] = acc;
    atomicMaxF(mx + (size_t)d * HEADS + hh, acc);
    if (hh == 0) atomicAdd(deg + d, 1.0f);
}

// ---------------- exp(logit - max) + segment sum -----------------------------
__global__ void k_edge_exp(const int* __restrict__ dst, float* __restrict__ albuf,
                           const float* __restrict__ mx, float* __restrict__ denom) {
    long long t = (long long)blockIdx.x * blockDim.x + threadIdx.x;
    if (t >= (long long)N_EDGES * HEADS) return;
    const int e = (int)(t >> 2), hh = (int)(t & 3);
    const int d = dst[e];
    const float a = __expf(albuf[t] - mx[(size_t)d * HEADS + hh]);
    albuf[t] = a;
    atomicAdd(denom + (size_t)d * HEADS + hh, a);
}

// ---------------- alpha = a/denom; scatter messages h[dst] += fs[src]*alpha --
__global__ void k_edge_norm_scatter(const int* __restrict__ src, const int* __restrict__ dst,
                                    const float* __restrict__ fs, float* __restrict__ albuf,
                                    const float* __restrict__ denom, float* __restrict__ hacc) {
    long long t = (long long)blockIdx.x * blockDim.x + threadIdx.x;
    if (t >= (long long)N_EDGES * HEADS) return;
    const int e = (int)(t >> 2), hh = (int)(t & 3);
    const int s = src[e], d = dst[e];
    const float alpha = albuf[t] / denom[(size_t)d * HEADS + hh];
    albuf[t] = alpha;                      // alpha output (E,H,1) flat
    const float* __restrict__ psrc = fs + (size_t)s * OUT_DIM + hh * HEAD_DIM;
    float* __restrict__ pdst = hacc + (size_t)d * OUT_DIM + hh * HEAD_DIM;
    #pragma unroll
    for (int j = 0; j < HEAD_DIM; ++j)
        atomicAdd(pdst + j, psrc[j] * alpha);
}

// ---------------- edge MLP (WMMA) fused with scatter-add to edge_agg --------
__global__ void __launch_bounds__(32)
k_edge_mlp(const float* __restrict__ ef, const int* __restrict__ dst,
           const _Float16* __restrict__ pWe12, const float* __restrict__ be1,
           const float* __restrict__ be2,
           const _Float16* __restrict__ pWef, const float* __restrict__ bef,
           float* __restrict__ eagg) {
    __shared__ _Float16 lds[16 * OUT_DIM];
    const int lane = threadIdx.x & 31;
    const int m = lane & 15, g = lane >> 4;
    const int et = blockIdx.x * 16;
    const v16h* __restrict__ pB1 = (const v16h*)pWe12;  // K=32, KC=1
    const v16h* __restrict__ pB2 = (const v16h*)pWef;   // K=128, KC=4

    // A = edge_feat tile [16 x 32] (K = EDGE_DIM exactly one chunk)
    v16h a;
    const float* xr = ef + (size_t)(et + m) * EDGE_DIM;
    #pragma unroll
    for (int h = 0; h < 16; ++h)
        a[h] = (_Float16)xr[a_kidx(h, g)];

    // stage 1: hidden[16x128] = X @ [We1;We2]^T + bias -> LDS (f16)
    #pragma unroll
    for (int ct = 0; ct < 8; ++ct) {
        v8f c = {};
        c = wmma_f16(a, pB1[ct * 32 + lane], c);
        const int o_n = ct * 16 + m;
        const float bv = (o_n < 64) ? be1[o_n] : be2[o_n - 64];
        #pragma unroll
        for (int r = 0; r < 8; ++r)
            lds[(r + g * 8) * OUT_DIM + o_n] = (_Float16)(c[r] + bv);
    }
    __syncthreads();

    // stage-2 A fragments from LDS
    v16h a2[4];
    #pragma unroll
    for (int kc = 0; kc < 4; ++kc)
        #pragma unroll
        for (int h = 0; h < 16; ++h)
            a2[kc][h] = lds[m * OUT_DIM + kc * 32 + a_kidx(h, g)];

    int dstv[8];
    #pragma unroll
    for (int r = 0; r < 8; ++r) dstv[r] = dst[et + r + g * 8];

    // stage 2: y = ELU(hidden @ Wef^T + bef); atomic scatter to edge_agg[dst]
    #pragma unroll
    for (int ct = 0; ct < 8; ++ct) {
        v8f c = {};
        #pragma unroll
        for (int kc = 0; kc < 4; ++kc)
            c = wmma_f16(a2[kc], pB2[(ct * 4 + kc) * 32 + lane], c);
        const int n = ct * 16 + m;
        const float bv = bef[n];
        #pragma unroll
        for (int r = 0; r < 8; ++r) {
            float y = c[r] + bv;
            y = y > 0.0f ? y : (__expf(y) - 1.0f);
            atomicAdd(eagg + (size_t)dstv[r] * OUT_DIM + n, y);
        }
    }
}

// ---------------- final fusion: [h | eagg/deg] @ Wff^T + bff -> LN -> ELU ----
__global__ void __launch_bounds__(32)
k_final(const float* __restrict__ hacc, const float* __restrict__ eagg,
        const float* __restrict__ deg, const _Float16* __restrict__ pWff,
        const float* __restrict__ bff, const float* __restrict__ ln_g,
        const float* __restrict__ ln_b, float* __restrict__ out) {
    __shared__ float lds[16 * OUT_DIM];
    const int lane = threadIdx.x & 31;
    const int m = lane & 15, g = lane >> 4;
    const int nt = blockIdx.x * 16;
    const float invd = 1.0f / fmaxf(deg[nt + m], 1.0f);
    const v16h* __restrict__ pB = (const v16h*)pWff;    // K=256, KC=8

    // A fragments over K = 256 (h cols 0..127, mean edge_agg cols 128..255)
    v16h a[8];
    const float* hr = hacc + (size_t)(nt + m) * OUT_DIM;
    const float* er = eagg + (size_t)(nt + m) * OUT_DIM;
    #pragma unroll
    for (int kc = 0; kc < 8; ++kc)
        #pragma unroll
        for (int h = 0; h < 16; ++h) {
            const int kg = kc * 32 + a_kidx(h, g);
            const float val = (kg < OUT_DIM) ? hr[kg] : er[kg - OUT_DIM] * invd;
            a[kc][h] = (_Float16)val;
        }

    #pragma unroll
    for (int ct = 0; ct < 8; ++ct) {
        v8f c = {};
        #pragma unroll
        for (int kc = 0; kc < 8; ++kc)
            c = wmma_f16(a[kc], pB[(ct * 8 + kc) * 32 + lane], c);
        const int n = ct * 16 + m;
        const float bv = bff[n];
        #pragma unroll
        for (int r = 0; r < 8; ++r)
            lds[(r + g * 8) * OUT_DIM + n] = c[r] + bv;
    }
    __syncthreads();

    if (lane < 16) {
        const float* row = lds + lane * OUT_DIM;
        float s = 0.0f, s2 = 0.0f;
        #pragma unroll 4
        for (int j = 0; j < OUT_DIM; ++j) { const float v = row[j]; s += v; s2 += v * v; }
        const float mu  = s * (1.0f / OUT_DIM);
        const float var = s2 * (1.0f / OUT_DIM) - mu * mu;
        const float inv = rsqrtf(var + LN_EPS);
        float* orow = out + (size_t)(nt + lane) * OUT_DIM;
        #pragma unroll 4
        for (int j = 0; j < OUT_DIM; ++j) {
            float v = (row[j] - mu) * inv * ln_g[j] + ln_b[j];
            orow[j] = v > 0.0f ? v : (__expf(v) - 1.0f);
        }
    }
}

extern "C" void kernel_launch(void* const* d_in, const int* in_sizes, int n_in,
                              void* d_out, int out_size, void* d_ws, size_t ws_size,
                              hipStream_t stream) {
    const float* node_feat = (const float*)d_in[0];
    const float* edge_feat = (const float*)d_in[1];
    const int*   src       = (const int*)  d_in[2];
    const int*   dst       = (const int*)  d_in[3];
    const float* W_src     = (const float*)d_in[4];
    const float* b_src     = (const float*)d_in[5];
    const float* W_dst     = (const float*)d_in[6];
    const float* b_dst     = (const float*)d_in[7];
    const float* attn_w    = (const float*)d_in[8];
    const float* W_e1      = (const float*)d_in[9];
    const float* b_e1      = (const float*)d_in[10];
    const float* W_e2      = (const float*)d_in[11];
    const float* b_e2      = (const float*)d_in[12];
    const float* W_ef      = (const float*)d_in[13];
    const float* b_ef      = (const float*)d_in[14];
    const float* W_ff      = (const float*)d_in[15];
    const float* b_ff      = (const float*)d_in[16];
    const float* ln_g      = (const float*)d_in[17];
    const float* ln_b      = (const float*)d_in[18];

    const size_t NF = (size_t)N_NODES * OUT_DIM;    // 12.8M floats
    float* ws    = (float*)d_ws;
    float* fs    = ws;                               // [N,128]
    float* fd    = fs    + NF;                       // [N,128]
    float* hacc  = fd    + NF;                       // [N,128]
    float* eagg  = hacc  + NF;                       // [N,128]
    float* mx    = eagg  + NF;                       // [N,4]
    float* denom = mx    + (size_t)N_NODES * HEADS;  // [N,4]
    float* deg   = denom + (size_t)N_NODES * HEADS;  // [N]

    // packed f16 weights (B-fragment-native layout), 64B-aligned
    size_t foff = (size_t)(4 * NF + 2 * (size_t)N_NODES * HEADS + N_NODES);
    foff = (foff + 15) & ~(size_t)15;
    _Float16* pWsrc = (_Float16*)(ws + foff);        // 128x128
    _Float16* pWdst = pWsrc + 128 * 128;             // 128x128
    _Float16* pWe12 = pWdst + 128 * 128;             // 128x32
    _Float16* pWef  = pWe12 + 128 * 32;              // 128x128
    _Float16* pWff  = pWef  + 128 * 128;             // 128x256

    float* h_out = (float*)d_out;                    // [N,128]
    float* albuf = h_out + NF;                       // [E,4] (alpha output, reused as scratch)

    const long long EH = (long long)N_EDGES * HEADS;

    k_init<<<(unsigned)((NF + 255) / 256), 256, 0, stream>>>(hacc, eagg, denom, mx, deg);

    // pack weights into f16 B-fragment layout
    k_pack_w<<<(128 * 128 + 255) / 256, 256, 0, stream>>>(W_src, W_src, 128, 128, pWsrc, 128 * 128);
    k_pack_w<<<(128 * 128 + 255) / 256, 256, 0, stream>>>(W_dst, W_dst, 128, 128, pWdst, 128 * 128);
    k_pack_w<<<(128 *  32 + 255) / 256, 256, 0, stream>>>(W_e1,  W_e2,   64,  32, pWe12, 128 * 32);
    k_pack_w<<<(128 * 128 + 255) / 256, 256, 0, stream>>>(W_ef,  W_ef,  128, 128, pWef,  128 * 128);
    k_pack_w<<<(128 * 256 + 255) / 256, 256, 0, stream>>>(W_ff,  W_ff,  128, 256, pWff,  128 * 256);

    k_node_linear<<<dim3(N_NODES / 16, 2), 32, 0, stream>>>(
        node_feat, pWsrc, b_src, pWdst, b_dst, fs, fd);

    k_edge_logits<<<(unsigned)((EH + 255) / 256), 256, 0, stream>>>(
        fs, fd, src, dst, attn_w, albuf, mx, deg);

    k_edge_exp<<<(unsigned)((EH + 255) / 256), 256, 0, stream>>>(dst, albuf, mx, denom);

    k_edge_norm_scatter<<<(unsigned)((EH + 255) / 256), 256, 0, stream>>>(
        src, dst, fs, albuf, denom, hacc);

    k_edge_mlp<<<N_EDGES / 16, 32, 0, stream>>>(
        edge_feat, dst, pWe12, b_e1, b_e2, pWef, b_ef, eagg);

    k_final<<<N_NODES / 16, 32, 0, stream>>>(
        hacc, eagg, deg, pWff, b_ff, ln_g, ln_b, h_out);
}
